// DynamicConvolution_10213432229938
// MI455X (gfx1250) — matrix operand order repaired and checked by reading, not compiled, not run on hardware
//
#include <hip/hip_runtime.h>
#include <hip/hip_bf16.h>
#include <math.h>

// ---------------------------------------------------------------------------
// Dynamic convolution for MI455X (gfx1250), wave32 + WMMA bf16.
//   out[b] = conv3x3(x[b], sum_k att[b,k]*W[k]) + att[b]@bias
// Implicit GEMM: [32 px] x [128 cout] tile per block, K = 9 taps * 128 cin.
// B tiles double-buffered via GLOBAL_LOAD_ASYNC_TO_LDS_B128 (ASYNCcnt).
// ---------------------------------------------------------------------------

typedef __attribute__((ext_vector_type(16))) __bf16 v16bf;
typedef __attribute__((ext_vector_type(8)))  float  v8f;

#define NB     16
#define C_IN   128
#define C_OUT  128
#define HW     128
#define NEXP   4
#define HID    64
#define TAPS   9

#if defined(__has_builtin)
#if __has_builtin(__builtin_amdgcn_global_load_async_to_lds_b128) && \
    __has_builtin(__builtin_amdgcn_s_wait_asynccnt)
#define USE_ASYNC_LDS 1
#endif
#endif

#if defined(USE_ASYNC_LDS)
// Exact parameter types per hipcc diagnostic:
//   param0: int __vector_size__(16) in AS(1) (global, printed as __device__)
//   param1: LDS-side pointer (AS(3)), same vector type
typedef int v4i_b128 __attribute__((vector_size(16)));
typedef __attribute__((address_space(1))) v4i_b128 gv4i;
typedef __attribute__((address_space(3))) v4i_b128 lv4i;
#endif

// f32 -> bf16 round-to-nearest-even (bit trick)
static __device__ __forceinline__ unsigned short f2bf(float f) {
  unsigned u = __float_as_uint(f);
  u = u + 0x7FFFu + ((u >> 16) & 1u);
  return (unsigned short)(u >> 16);
}

// ---------------------------------------------------------------------------
// 1) Global average pool: pooled[b*128+c] = mean over 128x128
// ---------------------------------------------------------------------------
__global__ __launch_bounds__(256) void pool_kernel(const float* __restrict__ x,
                                                   float* __restrict__ pooled) {
  int bc = blockIdx.x;                              // 0..2047 = b*128+c
  const float* p = x + (size_t)bc * (HW * HW);
  float s = 0.f;
  for (int i = threadIdx.x; i < HW * HW; i += 256) s += p[i];
  __shared__ float red[256];
  red[threadIdx.x] = s;
  __syncthreads();
  for (int off = 128; off > 0; off >>= 1) {
    if (threadIdx.x < off) red[threadIdx.x] += red[threadIdx.x + off];
    __syncthreads();
  }
  if (threadIdx.x == 0) pooled[bc] = red[0] * (1.0f / (HW * HW));
}

// ---------------------------------------------------------------------------
// 2) Attention MLP + softmax(/30) + blended bias (tiny, one block)
// ---------------------------------------------------------------------------
__global__ __launch_bounds__(256) void att_kernel(const float* __restrict__ pooled,
                                                  const float* __restrict__ w1,   // [64][128]
                                                  const float* __restrict__ w2,   // [4][64]
                                                  const float* __restrict__ bias, // [4][128]
                                                  float* __restrict__ att,        // [16][4]
                                                  float* __restrict__ aggb) {     // [16][128]
  __shared__ float sh_h[NB * HID];
  __shared__ float sh_att[NB * NEXP];
  int tid = threadIdx.x;
  for (int idx = tid; idx < NB * HID; idx += 256) {
    int b = idx >> 6, j = idx & 63;
    float s = 0.f;
    for (int c = 0; c < C_IN; ++c) s += pooled[b * C_IN + c] * w1[j * C_IN + c];
    sh_h[idx] = fmaxf(s, 0.f);
  }
  __syncthreads();
  if (tid < NB) {
    int b = tid;
    float lg[NEXP];
    float mx = -1e30f;
    for (int k = 0; k < NEXP; ++k) {
      float s = 0.f;
      for (int j = 0; j < HID; ++j) s += sh_h[b * HID + j] * w2[k * HID + j];
      lg[k] = s * (1.0f / 30.0f);
      mx = fmaxf(mx, lg[k]);
    }
    float den = 0.f;
    for (int k = 0; k < NEXP; ++k) { lg[k] = __expf(lg[k] - mx); den += lg[k]; }
    float inv = 1.0f / den;
    for (int k = 0; k < NEXP; ++k) {
      float a = lg[k] * inv;
      sh_att[b * NEXP + k] = a;
      att[b * NEXP + k] = a;
    }
  }
  __syncthreads();
  for (int idx = tid; idx < NB * C_OUT; idx += 256) {
    int b = idx >> 7, co = idx & 127;
    float s = 0.f;
    for (int k = 0; k < NEXP; ++k) s += sh_att[b * NEXP + k] * bias[k * C_OUT + co];
    aggb[idx] = s;
  }
}

// ---------------------------------------------------------------------------
// 3) Blend expert kernels into bf16, layout [b][tap][co][c] (GEMM-B friendly)
// ---------------------------------------------------------------------------
__global__ __launch_bounds__(256) void wagg_kernel(const float* __restrict__ att,    // [16][4]
                                                   const float* __restrict__ weight, // [4][128][128][3][3]
                                                   unsigned short* __restrict__ wbt) {
  int idx = blockIdx.x * 256 + threadIdx.x;  // 16*9*128*128 total
  int c   = idx & 127;
  int co  = (idx >> 7) & 127;
  int t   = idx >> 14;                       // b*9 + tap
  int b   = t / 9;
  int tap = t - b * 9;
  float a0 = att[b * 4 + 0], a1 = att[b * 4 + 1];
  float a2 = att[b * 4 + 2], a3 = att[b * 4 + 3];
  size_t wi = ((size_t)co * 128 + c) * 9 + tap;
  const size_t ks = (size_t)128 * 128 * 9;
  float s = a0 * weight[wi] + a1 * weight[wi + ks] +
            a2 * weight[wi + 2 * ks] + a3 * weight[wi + 3 * ks];
  // wbt[((b*9+tap)*128+co)*128+c] == wbt[idx]
  wbt[idx] = f2bf(s);
}

// ---------------------------------------------------------------------------
// 4) Implicit-GEMM conv with v_wmma_f32_16x16x32_bf16
//    block = (b, y, 32-px strip); 8 waves = 2(M) x 4(N); wave = 16x32 tile
//    36 K-steps: 4 channel-blocks x 9 taps; B tiles double-buffered async.
// ---------------------------------------------------------------------------
typedef union { v16bf v; unsigned u[8]; } Frag;

#define STRIDE_A 36   // halfs per A column (34x32 tile, padded)
#define STRIDE_B 40   // halfs per B row    (128x32 tile, 80B rows, 16B aligned)

__global__ __launch_bounds__(256, 2)
void conv_kernel(const float* __restrict__ x,
                 const unsigned short* __restrict__ wbt,   // [b][tap][co][c] bf16
                 const float* __restrict__ aggb,           // [16][128]
                 float* __restrict__ out) {
  __shared__ __align__(16) unsigned short lds_a[34 * STRIDE_A];      // [col][cc]
  __shared__ __align__(16) unsigned short lds_b[2][128 * STRIDE_B];  // [co][cc]

  int blk = blockIdx.x;
  int xt  = blk & 3;
  int y   = (blk >> 2) & 127;
  int b   = blk >> 9;
  int x0  = xt * 32;

  int tid   = threadIdx.x;
  int lane  = tid & 31;
  int w     = tid >> 5;
  int wm    = w & 1;        // M half (16 px)
  int wn    = w >> 1;       // N quarter (32 cout)
  int mlane = lane & 15;
  int hi    = lane >> 4;

  v8f acc0 = {}; v8f acc1 = {};
  const float* xb = x + (size_t)b * C_IN * HW * HW;

  // Issue (or synchronously perform) the copy of B tile `it` into buffer `bufi`.
  auto issue_b = [&](int it, int bufi) {
    int cbn  = it / 9;
    int tapn = it - cbn * 9;
    const unsigned short* wsrc =
        wbt + (size_t)(b * TAPS + tapn) * (C_OUT * C_IN) + cbn * 32;
#if defined(USE_ASYNC_LDS)
    // 512 x 16B transfers, 2 per thread (=> 2 async instructions per wave)
    for (int e = tid; e < 512; e += 256) {
      int co = e >> 2, g = e & 3;
      __builtin_amdgcn_global_load_async_to_lds_b128(
          (gv4i*)(wsrc + co * C_IN + g * 8),
          (lv4i*)(&lds_b[bufi][co * STRIDE_B + g * 8]),
          0, 0);
    }
#else
    for (int e = tid; e < 128 * 32; e += 256) {
      int co = e >> 5, cc = e & 31;
      lds_b[bufi][co * STRIDE_B + cc] = wsrc[co * C_IN + cc];
    }
#endif
  };

  issue_b(0, 0);
  int buf = 0;

  for (int it = 0; it < 36; ++it) {
    int cb  = it / 9;
    int tap = it - cb * 9;
    int ky  = tap / 3;
    int kx  = tap - ky * 3;

    if (kx == 0) {
      __syncthreads();  // prior readers of lds_a done
      int c0 = cb * 32;
      int gy = y + ky - 1;
      bool yin = ((unsigned)gy < 128u);
      // A tile: x[b, c0..c0+31, gy, x0-1..x0+32] -> bf16, transposed [col][cc]
      for (int e = tid; e < 34 * 32; e += 256) {
        int cc  = e / 34;
        int col = e - cc * 34;
        int gx  = x0 - 1 + col;
        float v = 0.f;
        if (yin && (unsigned)gx < 128u)
          v = xb[((size_t)(c0 + cc) * HW + gy) * HW + gx];
        lds_a[col * STRIDE_A + cc] = f2bf(v);
      }
    }

    __syncthreads();  // readers of lds_b[buf^1] (previous iter) done; lds_a ready
    if (it + 1 < 36) issue_b(it + 1, buf ^ 1);
#if defined(USE_ASYNC_LDS)
    // Retire the older (current-tile) transfers; keep next-tile ones in flight.
    if (it + 1 < 36) __builtin_amdgcn_s_wait_asynccnt(2);
    else             __builtin_amdgcn_s_wait_asynccnt(0);
#endif
    __syncthreads();  // lds_b[buf] visible block-wide

    // Build fragments (ISA 16-bit A layout; B from sparse-B analog)
    Frag af, bf0, bf1;
    int colb = wm * 16 + mlane + kx;   // column in lds_a halo window
    const unsigned* ap  = (const unsigned*)(lds_a + colb * STRIDE_A);
    const unsigned* bp0 = (const unsigned*)(lds_b[buf] + (wn * 32 + mlane) * STRIDE_B);
    const unsigned* bp1 = (const unsigned*)(lds_b[buf] + (wn * 32 + 16 + mlane) * STRIDE_B);
#pragma unroll
    for (int v = 0; v < 8; ++v) {
      int adw = hi * 4 + (v & 3) + ((v >> 2) << 3);  // K pair / 2
      af.u[v]  = ap[adw];
      int bdw = hi * 4 + (v & 3) + ((v >> 2) << 3);
      bf0.u[v] = bp0[bdw];
      bf1.u[v] = bp1[bdw];
    }
    acc0 = __builtin_amdgcn_wmma_f32_16x16x32_bf16(
        false, af.v, false, bf0.v, (short)0, acc0, false, false);
    acc1 = __builtin_amdgcn_wmma_f32_16x16x32_bf16(
        false, af.v, false, bf1.v, (short)0, acc1, false, false);

    buf ^= 1;
  }

  // Epilogue: D[M=r+hi*8][N=mlane] per lane, add blended bias.
  int co0 = wn * 32 + mlane;
  float bias0 = aggb[b * C_OUT + co0];
  float bias1 = aggb[b * C_OUT + co0 + 16];
  int pxb = x0 + wm * 16 + hi * 8;
  float* ob0 = out + (((size_t)b * C_OUT + co0) * HW + y) * HW + pxb;
  float* ob1 = ob0 + (size_t)16 * HW * HW;
#pragma unroll
  for (int r = 0; r < 8; ++r) {
    ob0[r] = acc0[r] + bias0;
    ob1[r] = acc1[r] + bias1;
  }
}

// ---------------------------------------------------------------------------
// Launch
// ---------------------------------------------------------------------------
extern "C" void kernel_launch(void* const* d_in, const int* in_sizes, int n_in,
                              void* d_out, int out_size, void* d_ws, size_t ws_size,
                              hipStream_t stream) {
  const float* x      = (const float*)d_in[0];  // [16,128,128,128]
  const float* att_w1 = (const float*)d_in[1];  // [64,128]
  const float* att_w2 = (const float*)d_in[2];  // [4,64]
  const float* weight = (const float*)d_in[3];  // [4,128,128,3,3]
  const float* bias   = (const float*)d_in[4];  // [4,128]
  float* out = (float*)d_out;

  // workspace layout
  char* ws = (char*)d_ws;
  unsigned short* wbt = (unsigned short*)ws;                 // 16*9*128*128 bf16
  size_t off = (size_t)NB * TAPS * C_OUT * C_IN * sizeof(unsigned short);
  float* pooled = (float*)(ws + off);           off += (size_t)NB * C_IN * sizeof(float);
  float* att    = (float*)(ws + off);           off += (size_t)NB * NEXP * sizeof(float);
  float* aggb   = (float*)(ws + off);

  pool_kernel<<<NB * C_IN, 256, 0, stream>>>(x, pooled);
  att_kernel<<<1, 256, 0, stream>>>(pooled, att_w1, att_w2, bias, att, aggb);
  wagg_kernel<<<(NB * TAPS * C_OUT * C_IN) / 256, 256, 0, stream>>>(att, weight, wbt);
  conv_kernel<<<NB * HW * (HW / 32), 256, 0, stream>>>(x, wbt, aggb, out);
}
